// MyVariableRNN_85976655331372
// MI455X (gfx1250) — matrix operand order repaired and checked by reading, not compiled, not run on hardware
//
#include <hip/hip_runtime.h>
#include <hip/hip_bf16.h>

// MI455X / gfx1250, wave32. Two kernels:
//  1) xg0_gemm: parallel WMMA GEMM  xg0[B*T,48] = x @ W_ih0^T + b_ih0  (bandwidth bound,
//     128 MB read + 48 MB write ~= 7.5 us at 23.3 TB/s). Written to d_ws (needs 48 MB).
//  2) gru5_pipeline: latency-bound scan. 32 workgroups x 5 waves; wave l owns layer l
//     (weights resident in ~150 VGPRs -> no VGPR-MSB churn), hidden tiles handed
//     between waves through double-buffered LDS, one workgroup barrier per pipeline
//     tick (s_barrier_signal/s_barrier_wait). Critical path per tick = ONE layer stage
//     (6 v_wmma_f32_16x16x32_f16 + gate VALU), not five.

typedef __attribute__((ext_vector_type(16))) _Float16 v16h;
typedef __attribute__((ext_vector_type(8)))  float    v8f;

#define BB 512
#define TT 512
#define DD 128
#define HH 16
#define GG 48   // 3*H
#define LL 5

__device__ __forceinline__ float sig_f(float x)  { return 1.0f / (1.0f + __expf(-x)); }
__device__ __forceinline__ float tanh_f(float x) { return 1.0f - 2.0f / (__expf(2.0f * x) + 1.0f); }

__device__ __forceinline__ v8f splat8(float v) {
  v8f c;
#pragma unroll
  for (int e = 0; e < 8; ++e) c[e] = v;
  return c;
}

// WMMA B operand (K x 16, K<=32) from row-major weight W[ngates][ldk]:
// B[k][n] = W[nb+n][kb+k]. Lane L: n = L&15, covers K = (L>>4)*8 + e (e<8), 16 + (L>>4)*8 + (e-8) (e>=8).
__device__ __forceinline__ v16h load_Btile(const float* W, int ldk, int nb,
                                           int kb, int kreal, int lane) {
  v16h b = {};
  const int n     = lane & 15;
  const int khalf = (lane >> 4) * 8;             // 0 or 8
  const float* row = W + (size_t)(nb + n) * ldk + kb;
#pragma unroll
  for (int e = 0; e < 16; ++e) {
    const int k = (e < 8) ? khalf + e : 16 + khalf + (e - 8);
    float v = 0.0f;
    if (k < kreal) v = row[k];                   // exec-masked; no OOB reads
    b[e] = (_Float16)v;
  }
  return b;
}

// A operand (16 x 32 f16) from per-lane f32 row pointer (lane's row = lane&15).
__device__ __forceinline__ v16h load_Atile_f32(const float* rowp, int kb, int lane) {
  const int hi = (lane >> 4) * 8;
  v16h a;
#pragma unroll
  for (int e = 0; e < 8; ++e) a[e]     = (_Float16)rowp[kb + hi + e];
#pragma unroll
  for (int e = 0; e < 8; ++e) a[8 + e] = (_Float16)rowp[kb + 16 + hi + e];
  return a;
}

// A operand (16 x 16 real, zero-padded to K=32) from LDS f16 tile hb[16][16].
__device__ __forceinline__ v16h load_Atile_lds(const _Float16* hb, int lane) {
  const int m  = lane & 15;
  const int hi = (lane >> 4) * 8;
  v16h a = {};
#pragma unroll
  for (int e = 0; e < 8; ++e) a[e] = hb[m * 16 + hi + e];  // 16B contiguous -> ds_load_b128
  return a;
}

// Store C-layout v8f hidden tile to LDS f16 [16][16] (row m, col n).
__device__ __forceinline__ void store_h_lds(_Float16* hb, const v8f h, int lane) {
  const int n  = lane & 15;
  const int mh = (lane >> 4) * 8;
#pragma unroll
  for (int e = 0; e < 8; ++e) hb[(mh + e) * 16 + n] = (_Float16)h[e];
}

__device__ __forceinline__ void gate_update(v8f& h, const v8f* ax, const v8f* ah) {
#pragma unroll
  for (int e = 0; e < 8; ++e) {
    const float r = sig_f(ax[0][e] + ah[0][e]);
    const float z = sig_f(ax[1][e] + ah[1][e]);
    const float n = tanh_f(ax[2][e] + r * ah[2][e]);
    h[e] = (1.0f - z) * n + z * h[e];
  }
}

// ============ Kernel 1: parallel layer-0 input projection ============
// xg0[r, g] = x[r, :] . W_ih0[g, :] + b_ih[0, g],  r = b*T + t  (x row order)
__global__ __launch_bounds__(256) void xg0_gemm(
    const float* __restrict__ x,       // [B*T, 128]
    const float* __restrict__ W_ih0,   // [48, 128]
    const float* __restrict__ b_ih,    // [5, 48] (layer 0 used)
    float* __restrict__ xg0)           // [B*T, 48]
{
  const int lane = threadIdx.x & 31;
  const int wave = threadIdx.x >> 5;                        // 8 waves / block
  const size_t r0 = ((size_t)blockIdx.x * 8 + wave) * 16;   // 16-row M tile
  const int n  = lane & 15;
  const int mh = (lane >> 4) * 8;

  v16h Bt[4][3];
#pragma unroll
  for (int kc = 0; kc < 4; ++kc)
#pragma unroll
    for (int nt = 0; nt < 3; ++nt)
      Bt[kc][nt] = load_Btile(W_ih0, DD, nt * 16, kc * 32, 32, lane);

  v8f acc[3];
#pragma unroll
  for (int nt = 0; nt < 3; ++nt) acc[nt] = splat8(b_ih[nt * 16 + n]);

  const float* rowp = x + (r0 + (lane & 15)) * DD;
#pragma unroll
  for (int kc = 0; kc < 4; ++kc) {
    const v16h a = load_Atile_f32(rowp, kc * 32, lane);
#pragma unroll
    for (int nt = 0; nt < 3; ++nt)
      acc[nt] = __builtin_amdgcn_wmma_f32_16x16x32_f16(
          false, a, false, Bt[kc][nt], (short)0, acc[nt], false, false);
  }

#pragma unroll
  for (int nt = 0; nt < 3; ++nt)
#pragma unroll
    for (int e = 0; e < 8; ++e)
      xg0[(r0 + mh + e) * GG + nt * 16 + n] = acc[nt][e];
}

// ============ Kernel 2: pipelined 5-layer GRU scan + FC head ============
__global__ __launch_bounds__(160) void gru5_pipeline(
    const float* __restrict__ xg0,     // [B*T, 48] from kernel 1
    const int*   __restrict__ lengths, // [B]
    const float* __restrict__ W_ihr,   // [4, 48, 16]
    const float* __restrict__ W_hh,    // [5, 48, 16]
    const float* __restrict__ b_ih,    // [5, 48]
    const float* __restrict__ b_hh,    // [5, 48]
    const float* __restrict__ W_fc,    // [2, 16]
    const float* __restrict__ b_fc,    // [2]
    float* __restrict__ out)           // [B, 2]
{
  const int lane = threadIdx.x & 31;
  const int l    = threadIdx.x >> 5;   // layer id 0..4, one wave per layer
  const int b0   = blockIdx.x * 16;    // batch tile
  const int n    = lane & 15;
  const int mh   = (lane >> 4) * 8;

  __shared__ _Float16 hbuf[LL][2][16][16];   // per-layer hidden, double-buffered by t parity
  __shared__ float    hfin[16][16];

  {
    _Float16* p = &hbuf[0][0][0][0];
    for (int i = threadIdx.x; i < LL * 2 * 256; i += 160) p[i] = (_Float16)0.0f;
  }
  __syncthreads();

  // ---- per-wave (per-layer) weights: 6 B-tiles = 48 VGPRs ----
  const v16h vzero = {};
  v16h Bx[3] = {vzero, vzero, vzero};
  if (l > 0) {
#pragma unroll
    for (int nt = 0; nt < 3; ++nt)
      Bx[nt] = load_Btile(W_ihr + (size_t)(l - 1) * GG * HH, HH, nt * 16, 0, 16, lane);
  }
  v16h Bh[3];
#pragma unroll
  for (int nt = 0; nt < 3; ++nt)
    Bh[nt] = load_Btile(W_hh + (size_t)l * GG * HH, HH, nt * 16, 0, 16, lane);

  float bih_[3], bhh_[3];
#pragma unroll
  for (int nt = 0; nt < 3; ++nt) {
    bih_[nt] = b_ih[l * GG + nt * 16 + n];
    bhh_[nt] = b_hh[l * GG + nt * 16 + n];
  }

  v8f h = splat8(0.0f);

  // =================== software pipeline over ticks ===================
  // Wave l processes t = tick - l. Producer(l) writes hbuf[l][t&1];
  // consumer(l+1) reads hbuf[l][(t-1)&1] the same tick -> no intra-tick race;
  // one barrier per tick orders across ticks.
#pragma unroll 1
  for (int tick = 0; tick < TT + LL - 1; ++tick) {
    const int t = tick - l;            // wave-uniform
    if (0 <= t && t < TT) {
      v8f ax[3], ah[3];
      if (l == 0) {
        // pre-biased gate tile from the parallel pre-pass
        const size_t rbase = ((size_t)(b0 + mh) * TT + t) * GG;
#pragma unroll
        for (int nt = 0; nt < 3; ++nt)
#pragma unroll
          for (int e = 0; e < 8; ++e)
            ax[nt][e] = xg0[rbase + (size_t)e * TT * GG + nt * 16 + n];
        if (t + 1 < TT)
          __builtin_prefetch(&xg0[rbase + GG], 0, 0);      // next timestep's tile
      } else {
        const v16h ap = load_Atile_lds(&hbuf[l - 1][t & 1][0][0], lane);
#pragma unroll
        for (int nt = 0; nt < 3; ++nt)
          ax[nt] = __builtin_amdgcn_wmma_f32_16x16x32_f16(
              false, ap, false, Bx[nt], (short)0, splat8(bih_[nt]), false, false);
      }
      const v16h ahh = load_Atile_lds(&hbuf[l][(t - 1) & 1][0][0], lane);
#pragma unroll
      for (int nt = 0; nt < 3; ++nt)
        ah[nt] = __builtin_amdgcn_wmma_f32_16x16x32_f16(
            false, ahh, false, Bh[nt], (short)0, splat8(bhh_[nt]), false, false);
      gate_update(h, ax, ah);
      store_h_lds(&hbuf[l][t & 1][0][0], h, lane);
    }
    __syncthreads();
  }

  // =================== FC head + length mask ===================
  if (l == LL - 1) {
#pragma unroll
    for (int e = 0; e < 8; ++e) hfin[mh + e][n] = h[e];
  }
  __syncthreads();

  if (threadIdx.x < 16) {
    const int b = b0 + (int)threadIdx.x;
    float a0 = b_fc[0], a1 = b_fc[1];
#pragma unroll
    for (int k = 0; k < HH; ++k) {
      const float hv = hfin[threadIdx.x][k];
      a0 += hv * W_fc[k];
      a1 += hv * W_fc[HH + k];
    }
    const bool valid = (lengths[b] == TT);
    out[b * 2 + 0] = valid ? a0 : 0.0f;
    out[b * 2 + 1] = valid ? a1 : 0.0f;
  }
}

extern "C" void kernel_launch(void* const* d_in, const int* in_sizes, int n_in,
                              void* d_out, int out_size, void* d_ws, size_t ws_size,
                              hipStream_t stream) {
  (void)in_sizes; (void)n_in; (void)out_size; (void)ws_size;
  const float* x       = (const float*)d_in[0];
  const int*   lengths = (const int*)  d_in[1];
  const float* W_ih0   = (const float*)d_in[2];
  const float* W_ihr   = (const float*)d_in[3];
  const float* W_hh    = (const float*)d_in[4];
  const float* b_ih    = (const float*)d_in[5];
  const float* b_hh    = (const float*)d_in[6];
  const float* W_fc    = (const float*)d_in[7];
  const float* b_fc    = (const float*)d_in[8];
  float* out = (float*)d_out;
  float* xg0 = (float*)d_ws;           // needs B*T*48*4 = 48 MB scratch

  // 1) parallel layer-0 projection: 2048 blocks x 8 waves, one 16x48 tile each
  xg0_gemm<<<dim3((BB * TT) / (16 * 8)), dim3(256), 0, stream>>>(x, W_ih0, b_ih, xg0);

  // 2) pipelined scan: 32 workgroups x 5 waves (one per layer)
  gru5_pipeline<<<dim3(BB / 16), dim3(160), 0, stream>>>(
      xg0, lengths, W_ihr, W_hh, b_ih, b_hh, W_fc, b_fc, out);
}
